// MultiHeadAttention_79611513799233
// MI455X (gfx1250) — matrix operand order repaired and checked by reading, not compiled, not run on hardware
//
#include <hip/hip_runtime.h>
#include <hip/hip_bf16.h>

// MI455X / gfx1250 multi-head attention.
// - all matmuls via v_wmma_f32_16x16x32_f16 (fp32 accumulate)
// - 64x64 wave tiles in the projection GEMMs, #pragma unroll 2 K-loop
// - attention: K/V blocks double-buffered in LDS via
//   GLOBAL_LOAD_ASYNC_TO_LDS_B128 with partial s_wait_asynccnt overlap.

typedef _Float16 h16 __attribute__((ext_vector_type(16)));
typedef _Float16 h8  __attribute__((ext_vector_type(8)));
typedef float    f8  __attribute__((ext_vector_type(8)));
typedef int      v4i __attribute__((ext_vector_type(4)));

typedef void __attribute__((address_space(1))) as1_void;
typedef void __attribute__((address_space(3))) as3_void;
typedef v4i  __attribute__((address_space(1))) as1_v4i;
typedef v4i  __attribute__((address_space(3))) as3_v4i;

#if defined(__has_builtin)
#if __has_builtin(__builtin_amdgcn_global_load_async_to_lds_b128) && \
    __has_builtin(__builtin_amdgcn_s_wait_asynccnt)
#define USE_ASYNC_LDS 1
#endif
#endif
#ifndef USE_ASYNC_LDS
#define USE_ASYNC_LDS 0
#endif

#define B_  2
#define S_  2048
#define D_  1024
#define H_  16
#define HD_ 64

union H16u { h16 v; h8 h[2]; };

__device__ __forceinline__ f8 wmma_f16(h16 a, h16 b, f8 c) {
  return __builtin_amdgcn_wmma_f32_16x16x32_f16(false, a, false, b, (short)0, c,
                                                false, false);
}

// A-operand fragment (16x32, row-major source, K contiguous).
__device__ __forceinline__ h16 load_a_frag(const _Float16* A, int lda, int row0,
                                           int kk, int lane) {
  int row = row0 + (lane & 15);
  int kb  = kk + ((lane >> 4) << 3);
  const _Float16* p = A + (size_t)row * lda + kb;
  H16u u;
  u.h[0] = *(const h8*)(p);
  u.h[1] = *(const h8*)(p + 16);
  return u.v;
}

// B-operand fragment (32x16): lane holds column n0+(lane&15), K in table order.
__device__ __forceinline__ h16 load_b_frag(const _Float16* Bt, int ldb, int kk,
                                           int n0, int lane) {
  int n  = n0 + (lane & 15);
  int kb = kk + ((lane >> 4) << 3);
  h16 r;
#pragma unroll
  for (int s = 0; s < 8; s++) {
    r[s]     = Bt[(size_t)(kb + s) * ldb + n];
    r[s + 8] = Bt[(size_t)(kb + 16 + s) * ldb + n];
  }
  return r;
}

// 16-byte global -> LDS stage (async on CDNA5 if builtin available).
__device__ __forceinline__ void stage16(const _Float16* g, _Float16* l) {
#if USE_ASYNC_LDS
  __builtin_amdgcn_global_load_async_to_lds_b128(
      (as1_v4i*)(as1_void*)(void*)g, (as3_v4i*)(as3_void*)(void*)l, 0, 0);
#else
  *(h8*)l = *(const h8*)g;
#endif
}
template <int N> __device__ __forceinline__ void stage_wait() {
#if USE_ASYNC_LDS
  __builtin_amdgcn_s_wait_asynccnt(N);
#endif
}

// ---------------------------------------------------------------------------
// Phase 0: conversions / weight transposes
// ---------------------------------------------------------------------------
__global__ void cvt_f32_to_f16(const float* __restrict__ src,
                               _Float16* __restrict__ dst, int n) {
  int i = blockIdx.x * blockDim.x + threadIdx.x;
  if (i < n) dst[i] = (_Float16)src[i];
}

// Wt[k][n] = W[n][k] so WMMA B operand reads are contiguous.
__global__ void transpose_w_f16(const float* __restrict__ W,
                                _Float16* __restrict__ Wt) {
  int i = blockIdx.x * blockDim.x + threadIdx.x;
  int n = i & (D_ - 1);
  int k = i >> 10;
  Wt[(size_t)k * D_ + n] = (_Float16)W[(size_t)n * D_ + k];
}

// ---------------------------------------------------------------------------
// Phase 1: fused QKV projection. One wave = 64x64 tile (4x4 WMMA).
// ---------------------------------------------------------------------------
__global__ __launch_bounds__(128) void qkv_proj(const _Float16* __restrict__ xh,
                                                const _Float16* __restrict__ WT,
                                                _Float16* __restrict__ Qs,
                                                _Float16* __restrict__ KhT,
                                                _Float16* __restrict__ Vh) {
  int lane = threadIdx.x & 31;
  int wave = threadIdx.x >> 5;
  int gw   = blockIdx.x * 4 + wave;          // 0 .. 3*64*16-1
  int mat  = gw / (64 * 16);
  int rem  = gw % (64 * 16);
  int m0   = (rem & 63) * 64;                // row tile in [0,4096)
  int n0   = (rem >> 6) * 64;                // col tile in [0,1024)
  const _Float16* Bt = WT + (size_t)mat * D_ * D_;

  f8 acc[4][4] = {};
#pragma unroll 2
  for (int kk = 0; kk < D_; kk += 32) {
    // next K-panel prefetch; one-past-the-end still lands inside workspace
    __builtin_prefetch(Bt + (size_t)(kk + 32 + lane) * D_ + n0, 0, 0);
    h16 a[4], b[4];
#pragma unroll
    for (int i = 0; i < 4; i++) a[i] = load_a_frag(xh, D_, m0 + 16 * i, kk, lane);
#pragma unroll
    for (int j = 0; j < 4; j++) b[j] = load_b_frag(Bt, D_, kk, n0 + 16 * j, lane);
#pragma unroll
    for (int i = 0; i < 4; i++)
#pragma unroll
      for (int j = 0; j < 4; j++) acc[i][j] = wmma_f16(a[i], b[j], acc[i][j]);
  }

  int cl = lane & 15;
  int rh = (lane >> 4) << 3;
#pragma unroll
  for (int ti = 0; ti < 4; ti++)
#pragma unroll
    for (int tj = 0; tj < 4; tj++)
#pragma unroll
      for (int v = 0; v < 8; v++) {
        int row = m0 + ti * 16 + rh + v;     // b*S + s
        int col = n0 + tj * 16 + cl;         // h*HD + hd
        float val = acc[ti][tj][v];
        int b = row >> 11, s = row & (S_ - 1);
        int h = col >> 6,  hd = col & 63;
        if (mat == 0)
          Qs[(((size_t)(b * H_ + h) * S_ + s) * HD_) + hd] =
              (_Float16)(val * 0.125f);      // fold 1/sqrt(HD)
        else if (mat == 1)
          KhT[(((size_t)(b * H_ + h) * HD_ + hd) * S_) + s] = (_Float16)val;
        else
          Vh[(((size_t)(b * H_ + h) * S_ + s) * HD_) + hd] = (_Float16)val;
      }
}

// ---------------------------------------------------------------------------
// Phase 2: attention. Block = one (b,h) x 64 query rows; wave = 16 rows.
// K/V blocks double-buffered in LDS (async), shared by the 4 waves.
// ---------------------------------------------------------------------------
__global__ __launch_bounds__(128) void attn(const _Float16* __restrict__ Qs,
                                            const _Float16* __restrict__ KhT,
                                            const _Float16* __restrict__ Vh,
                                            _Float16* __restrict__ ctxh) {
  __shared__ __align__(16) _Float16 sK[2][64 * 32];   // [hd][s-in-block]
  __shared__ __align__(16) _Float16 sV[2][32 * 64];   // [s-in-block][hd]
  __shared__ __align__(16) _Float16 sP[4][16 * 32];   // per-wave P tile
  int t    = threadIdx.x;
  int lane = t & 31;
  int wave = t >> 5;
  int bh   = blockIdx.x >> 5;                // 0..31 = b*H + h
  int qc   = blockIdx.x & 31;                // 64-row query chunk

  const _Float16* Q  = Qs  + ((size_t)bh * S_ + qc * 64 + wave * 16) * HD_;
  const _Float16* KT = KhT + (size_t)bh * HD_ * S_;
  const _Float16* V  = Vh  + (size_t)bh * S_ * HD_;

  // stage K block (64 x 32 halves): 256 x 16B chunks, 2 per thread
  auto stageK = [&](int j, int buf) {
#pragma unroll
    for (int u = 0; u < 2; u++) {
      int c  = t * 2 + u;
      int hd = c >> 2, off = (c & 3) * 8;
      stage16(KT + (size_t)hd * S_ + j + off, sK[buf] + hd * 32 + off);
    }
  };
  // stage V block (32 x 64 halves): 256 x 16B chunks, 2 per thread
  auto stageV = [&](int j, int buf) {
#pragma unroll
    for (int u = 0; u < 2; u++) {
      int c   = t * 2 + u;
      int row = c >> 3, off = (c & 7) * 8;
      stage16(V + (size_t)(j + row) * HD_ + off, sV[buf] + row * 64 + off);
    }
  };

  h16 qa0 = load_a_frag(Q, HD_, 0, 0,  lane);   // Q pre-scaled by 1/8
  h16 qa1 = load_a_frag(Q, HD_, 0, 32, lane);

  int cl = lane & 15;
  int rh = (lane >> 4) << 3;

  // ---- pass 1: row max (double-buffered K staging) ----
  float m[8];
#pragma unroll
  for (int v = 0; v < 8; v++) m[v] = -3.0e38f;
  stageK(0, 0);
  for (int j = 0; j < S_; j += 32) {
    int buf = (j >> 5) & 1;
    if (j + 32 < S_) {                 // issue next block, then partial wait
      stageK(j + 32, buf ^ 1);
      stage_wait<2>();
    } else {
      stage_wait<0>();
    }
    __syncthreads();
    f8 s0 = {}, s1 = {};
    s0 = wmma_f16(qa0, load_b_frag(sK[buf], 32, 0,  0,  lane), s0);
    s0 = wmma_f16(qa1, load_b_frag(sK[buf], 32, 32, 0,  lane), s0);
    s1 = wmma_f16(qa0, load_b_frag(sK[buf], 32, 0,  16, lane), s1);
    s1 = wmma_f16(qa1, load_b_frag(sK[buf], 32, 32, 16, lane), s1);
#pragma unroll
    for (int v = 0; v < 8; v++) m[v] = fmaxf(m[v], fmaxf(s0[v], s1[v]));
    __syncthreads();
  }
#pragma unroll
  for (int v = 0; v < 8; v++)
    for (int xm = 1; xm < 16; xm <<= 1)
      m[v] = fmaxf(m[v], __shfl_xor(m[v], xm, 32));

  // ---- pass 2: exp / row-sum / ctx (double-buffered K+V staging) ----
  float l[8];
#pragma unroll
  for (int v = 0; v < 8; v++) l[v] = 0.0f;
  f8 c4[4] = {};
  _Float16* Pt = sP[wave];

  stageK(0, 0);
  stageV(0, 0);
  for (int j = 0; j < S_; j += 32) {
    int buf = (j >> 5) & 1;
    if (j + 32 < S_) {
      stageK(j + 32, buf ^ 1);
      stageV(j + 32, buf ^ 1);
      stage_wait<4>();
    } else {
      stage_wait<0>();
    }
    __syncthreads();

    f8 s0 = {}, s1 = {};
    s0 = wmma_f16(qa0, load_b_frag(sK[buf], 32, 0,  0,  lane), s0);
    s0 = wmma_f16(qa1, load_b_frag(sK[buf], 32, 32, 0,  lane), s0);
    s1 = wmma_f16(qa0, load_b_frag(sK[buf], 32, 0,  16, lane), s1);
    s1 = wmma_f16(qa1, load_b_frag(sK[buf], 32, 32, 16, lane), s1);
#pragma unroll
    for (int v = 0; v < 8; v++) {
      float p0 = exp2f((s0[v] - m[v]) * 1.44269504f);
      float p1 = exp2f((s1[v] - m[v]) * 1.44269504f);
      l[v] += p0 + p1;
      Pt[(rh + v) * 32 + cl]      = (_Float16)p0;   // C-layout -> LDS
      Pt[(rh + v) * 32 + cl + 16] = (_Float16)p1;
    }
    __syncthreads();                                // P ready; K reads done
    h16 pa = load_a_frag(Pt, 32, 0, 0, lane);       // A-layout re-read
#pragma unroll
    for (int tt = 0; tt < 4; tt++)
      c4[tt] = wmma_f16(pa, load_b_frag(sV[buf], 64, 0, tt * 16, lane), c4[tt]);
    __syncthreads();                                // V/P reads done
  }

#pragma unroll
  for (int v = 0; v < 8; v++) {
    for (int xm = 1; xm < 16; xm <<= 1) l[v] += __shfl_xor(l[v], xm, 32);
    l[v] = 1.0f / l[v];
  }

  int b = bh >> 4, h = bh & 15;
#pragma unroll
  for (int tt = 0; tt < 4; tt++)
#pragma unroll
    for (int v = 0; v < 8; v++) {
      int s = qc * 64 + wave * 16 + rh + v;
      int d = h * 64 + tt * 16 + cl;
      ctxh[((size_t)(b * S_ + s)) * D_ + d] = (_Float16)(c4[tt][v] * l[v]);
    }
}

// ---------------------------------------------------------------------------
// Phase 3: output projection + bias, fp32 out. 64x64 wave tiles.
// ---------------------------------------------------------------------------
__global__ __launch_bounds__(128) void out_proj(const _Float16* __restrict__ ctxh,
                                                const _Float16* __restrict__ WoT,
                                                const float* __restrict__ bo,
                                                float* __restrict__ out) {
  int lane = threadIdx.x & 31;
  int wave = threadIdx.x >> 5;
  int gw   = blockIdx.x * 4 + wave;          // 0..1023
  int m0   = (gw & 63) * 64;
  int n0   = (gw >> 6) * 64;

  f8 acc[4][4] = {};
#pragma unroll 2
  for (int kk = 0; kk < D_; kk += 32) {
    __builtin_prefetch(WoT + (size_t)(kk + 32 + lane) * D_ + n0, 0, 0);
    h16 a[4], b[4];
#pragma unroll
    for (int i = 0; i < 4; i++) a[i] = load_a_frag(ctxh, D_, m0 + 16 * i, kk, lane);
#pragma unroll
    for (int j = 0; j < 4; j++) b[j] = load_b_frag(WoT, D_, kk, n0 + 16 * j, lane);
#pragma unroll
    for (int i = 0; i < 4; i++)
#pragma unroll
      for (int j = 0; j < 4; j++) acc[i][j] = wmma_f16(a[i], b[j], acc[i][j]);
  }

  int cl = lane & 15;
  int rh = (lane >> 4) << 3;
#pragma unroll
  for (int ti = 0; ti < 4; ti++)
#pragma unroll
    for (int tj = 0; tj < 4; tj++)
#pragma unroll
      for (int v = 0; v < 8; v++) {
        int row = m0 + ti * 16 + rh + v;
        int col = n0 + tj * 16 + cl;
        out[(size_t)row * D_ + col] = acc[ti][tj][v] + bo[col];
      }
}

// ---------------------------------------------------------------------------
// Host-side launcher
// ---------------------------------------------------------------------------
extern "C" void kernel_launch(void* const* d_in, const int* in_sizes, int n_in,
                              void* d_out, int out_size, void* d_ws, size_t ws_size,
                              hipStream_t stream) {
  (void)in_sizes; (void)n_in; (void)out_size; (void)ws_size;
  const float* x  = (const float*)d_in[0];
  const float* Wq = (const float*)d_in[1];
  const float* Wk = (const float*)d_in[2];
  const float* Wv = (const float*)d_in[3];
  const float* Wo = (const float*)d_in[4];
  const float* bo = (const float*)d_in[5];

  const size_t MB1 = (size_t)D_ * D_;        // 1 Mi elements
  _Float16* ws   = (_Float16*)d_ws;
  _Float16* xh   = ws;                       // 4 Mi
  _Float16* WT   = ws + 4  * MB1;            // WqT,WkT,WvT
  _Float16* WoT  = ws + 7  * MB1;
  _Float16* Qsc  = ws + 8  * MB1;            // Q*1/8, [B,H,S,HD]
  _Float16* KhT  = ws + 12 * MB1;            // [B,H,HD,S]
  _Float16* Vh   = ws + 16 * MB1;            // [B,H,S,HD]
  _Float16* ctxh = ws + 20 * MB1;            // [B,S,D]

  const int nX = B_ * S_ * D_;
  cvt_f32_to_f16<<<nX / 256, 256, 0, stream>>>(x, xh, nX);
  transpose_w_f16<<<(int)(MB1 / 256), 256, 0, stream>>>(Wq, WT);
  transpose_w_f16<<<(int)(MB1 / 256), 256, 0, stream>>>(Wk, WT + MB1);
  transpose_w_f16<<<(int)(MB1 / 256), 256, 0, stream>>>(Wv, WT + 2 * MB1);
  transpose_w_f16<<<(int)(MB1 / 256), 256, 0, stream>>>(Wo, WoT);

  qkv_proj<<<768, 128, 0, stream>>>(xh, WT, Qsc, KhT, Vh);
  attn<<<1024, 128, 0, stream>>>(Qsc, KhT, Vh, ctxh);
  out_proj<<<256, 128, 0, stream>>>(ctxh, WoT, bo, (float*)d_out);
}
